// TransformerBlock_55937654063307
// MI455X (gfx1250) — matrix operand order repaired and checked by reading, not compiled, not run on hardware
//
#include <hip/hip_runtime.h>
#include <stdint.h>

// Problem constants (B=2, T=2048, D=1024, H=16, HS=64)
#define DMODEL 1024
#define SEQ    2048
#define NBATCH 2
#define NHEAD  16
#define HSZ    64
#define NROWS  (NBATCH * SEQ)   // 4096
#define DFF    (4 * DMODEL)     // 4096

typedef __bf16 bf16;
typedef __attribute__((ext_vector_type(16))) __bf16 v16bf;
typedef __attribute__((ext_vector_type(8)))  float  v8f;

// ---------------------------------------------------------------------------
// bf16 conversion (round-to-nearest-even)
__device__ __forceinline__ bf16 to_bf16(float f) {
  union { float f; uint32_t u; } v; v.f = f;
  uint32_t r = v.u + 0x7fffu + ((v.u >> 16) & 1u);
  uint16_t h = (uint16_t)(r >> 16);
  bf16 b; __builtin_memcpy(&b, &h, 2);
  return b;
}

// K-index patterns per CDNA5 ISA 7.12.2 (wave32, 16-bit operands).
// A 16x32: lanes 0-15 hold K{0..7,16..23}, lanes 16-31 hold K{8..15,24..31}
__device__ __forceinline__ int kpatA(int j, int hi) {
  return (j < 4) ? (hi * 8 + 2 * j) : (16 + hi * 8 + 2 * (j - 4));
}
// B 32x16: lanes 0-15 hold K{0..7,16..23}, lanes 16-31 hold K{8..15,24..31}
__device__ __forceinline__ int kpatB(int j, int hi) {
  return (j < 4) ? (hi * 8 + 2 * j) : (16 + hi * 8 + 2 * (j - 4));
}

// A fragment from row-major [M][ld] storage, 16x32 tile at (row0, k0).
// Per lane: two contiguous 8-element (16B) runs -> vectorizes to ds_load_b128.
__device__ __forceinline__ v16bf load_a_rm(const bf16* src, int ld, int row0,
                                           int k0, int lane) {
  const bf16* p = src + (size_t)(row0 + (lane & 15)) * ld + k0;
  int hi = (lane >> 4) & 1;
  v16bf f;
#pragma unroll
  for (int j = 0; j < 8; ++j) {
    int kk = kpatA(j, hi);
    f[2 * j]     = p[kk];
    f[2 * j + 1] = p[kk + 1];
  }
  return f;
}

// B fragment where B[k][n] = S[n][k] for row-major S [N][ld].
// Per lane: two contiguous 8-element (16B) runs -> 2x ds_load_b128.
__device__ __forceinline__ v16bf load_b_T(const bf16* src, int ld, int col0,
                                          int k0, int lane) {
  const bf16* p = src + (size_t)(col0 + (lane & 15)) * ld + k0;
  int hi = (lane >> 4) & 1;
  v16bf f;
#pragma unroll
  for (int j = 0; j < 8; ++j) {
    int kk = kpatB(j, hi);
    f[2 * j]     = p[kk];
    f[2 * j + 1] = p[kk + 1];
  }
  return f;
}

__device__ __forceinline__ v8f wmma_bf16(v16bf a, v16bf b, v8f c) {
  return __builtin_amdgcn_wmma_f32_16x16x32_bf16(false, a, false, b,
                                                 (short)0, c, false, false);
}

// ---------------------------------------------------------------------------
// LayerNorm: one 256-thread block per row of D=1024, outputs bf16.
__global__ void __launch_bounds__(256, 2)
ln_kernel(const float* __restrict__ x, const float* __restrict__ g,
          const float* __restrict__ be, bf16* __restrict__ out) {
  int row = blockIdx.x;
  int tid = threadIdx.x;
  const float* xr = x + (size_t)row * DMODEL;
  float v[4];
  float s = 0.f, ss = 0.f;
#pragma unroll
  for (int i = 0; i < 4; ++i) {
    v[i] = xr[tid + i * 256];
    s += v[i];
    ss += v[i] * v[i];
  }
  __shared__ float rs[256], rss[256];
  rs[tid] = s; rss[tid] = ss;
  __syncthreads();
  for (int off = 128; off > 0; off >>= 1) {
    if (tid < off) { rs[tid] += rs[tid + off]; rss[tid] += rss[tid + off]; }
    __syncthreads();
  }
  float mean = rs[0] * (1.0f / DMODEL);
  float var  = rss[0] * (1.0f / DMODEL) - mean * mean;
  float inv  = rsqrtf(var + 1e-5f);
  bf16* orow = out + (size_t)row * DMODEL;
#pragma unroll
  for (int i = 0; i < 4; ++i) {
    int c = tid + i * 256;
    orow[c] = to_bf16((v[i] - mean) * inv * g[c] + be[c]);
  }
}

// ---------------------------------------------------------------------------
// Tiled bf16 WMMA GEMM: C[M,N] = A[M,K](bf16) * B[K,N](fp32 weights, converted)
// Block tile 128x128, 8 waves arranged 4(M) x 2(N): each wave owns 32x64.
// Double-buffered LDS; B tile stored TRANSPOSED (BsT[n][k]) so B-fragment
// loads are lane-contiguous ds_load_b128 instead of 16 strided u16 loads.
// BMODE 0: B row-major [K][N];  BMODE 1: head-blocked [H][K][64] (Wq/Wk/Wv).
#define F_BIAS  1
#define F_RELU  2
#define F_RESID 4
#define F_OUTF  8
#define F_OUTB  16

template <int BMODE, int FLAGS>
__global__ void __launch_bounds__(256, 2)
gemm_kernel(const bf16* __restrict__ A, const float* __restrict__ Bw,
            const float* __restrict__ bias,
            const float* __restrict__ resid,
            float* __restrict__ outF, bf16* __restrict__ outB,
            int M, int N, int K) {
  __shared__ bf16 As[2][128 * 32];
  __shared__ bf16 BsT[2][128 * 32];   // [n][k] transposed
  int tid  = threadIdx.x;
  int lane = tid & 31;
  int w    = tid >> 5;
  int hi   = (lane >> 4) & 1;
  int m0 = blockIdx.y * 128;
  int n0 = blockIdx.x * 128;
  int mBase = (w >> 1) * 32;
  int nBase = (w & 1) * 64;

  v8f zero = {};
  v8f acc[2][4];
#pragma unroll
  for (int i = 0; i < 2; ++i)
#pragma unroll
    for (int j = 0; j < 4; ++j) acc[i][j] = zero;

  // Per-thread staging: 16 contiguous global elements each for A and B.
  int aRow = tid >> 1, aCol = (tid & 1) * 16;
  int bRow = tid >> 3, bCol = (tid & 7) * 16;

  auto stage = [&](int k0, int bi) {
    const bf16* ag = A + (size_t)(m0 + aRow) * K + k0 + aCol;
    bf16* as = &As[bi][aRow * 32 + aCol];
#pragma unroll
    for (int j = 0; j < 16; ++j) as[j] = ag[j];

    bf16* bs = &BsT[bi][0];
    if (BMODE == 0) {
      const float* bg = Bw + (size_t)(k0 + bRow) * N + n0 + bCol;
#pragma unroll
      for (int j = 0; j < 16; ++j)
        bs[(bCol + j) * 32 + bRow] = to_bf16(bg[j]);   // transpose on store
    } else {
      int n = n0 + bCol;            // 16-col run stays inside one head
      int h = n >> 6, e = n & 63;
      const float* bg = Bw + ((size_t)h * K + (k0 + bRow)) * 64 + e;
#pragma unroll
      for (int j = 0; j < 16; ++j)
        bs[(bCol + j) * 32 + bRow] = to_bf16(bg[j]);   // transpose on store
    }
  };

  stage(0, 0);
  __syncthreads();

  int nk = K / 32;
  for (int kt = 0; kt < nk; ++kt) {
    int cur = kt & 1;
    if (kt + 1 < nk) stage((kt + 1) * 32, cur ^ 1);

    v16bf a0 = load_a_rm(As[cur], 32, mBase, 0, lane);
    v16bf a1 = load_a_rm(As[cur], 32, mBase + 16, 0, lane);
#pragma unroll
    for (int nb = 0; nb < 4; ++nb) {
      v16bf bb = load_b_T(BsT[cur], 32, nBase + nb * 16, 0, lane);
      acc[0][nb] = wmma_bf16(a0, bb, acc[0][nb]);
      acc[1][nb] = wmma_bf16(a1, bb, acc[1][nb]);
    }
    __syncthreads();
  }

  // Epilogue: C layout row = r + 8*hi, col = lane&15 (ISA 7.12.2)
#pragma unroll
  for (int mb = 0; mb < 2; ++mb)
#pragma unroll
    for (int nb = 0; nb < 4; ++nb)
#pragma unroll
      for (int r = 0; r < 8; ++r) {
        int row = m0 + mBase + mb * 16 + r + 8 * hi;
        int col = n0 + nBase + nb * 16 + (lane & 15);
        float vv = acc[mb][nb][r];
        if (FLAGS & F_BIAS) vv += bias[col];
        if (FLAGS & F_RELU) vv = fmaxf(vv, 0.0f);
        size_t idx = (size_t)row * N + col;
        if (FLAGS & F_RESID) vv += resid[idx];
        if (FLAGS & F_OUTF) outF[idx] = vv;
        if (FLAGS & F_OUTB) outB[idx] = to_bf16(vv);
      }
}

// ---------------------------------------------------------------------------
// Flash attention, causal. q/k/v/o are bf16 [B*T][H*HS] row-major.
// Grid: (T/128, B*H). Each of 8 waves owns a 16-row query strip.
// V tile stored TRANSPOSED (VtT[e][s]) so P·V B-fragments are contiguous.
__global__ void __launch_bounds__(256, 1)
attn_kernel(const bf16* __restrict__ q, const bf16* __restrict__ k,
            const bf16* __restrict__ v, bf16* __restrict__ o) {
  __shared__ bf16 Kt[64 * 64];        // [s][e]
  __shared__ bf16 VtT[64 * 64];       // [e][s] transposed
  __shared__ bf16 Pt[8][16 * 64];     // per-wave P staging (C-layout -> A-frag)
  int tid  = threadIdx.x;
  int lane = tid & 31;
  int w    = tid >> 5;
  int hi   = (lane >> 4) & 1;
  int b = blockIdx.y / NHEAD;
  int h = blockIdx.y % NHEAD;
  int q0 = blockIdx.x * 128;
  int rowBase = q0 + w * 16;

  const bf16* qp = q + (size_t)b * SEQ * DMODEL + h * HSZ;
  const bf16* kp = k + (size_t)b * SEQ * DMODEL + h * HSZ;
  const bf16* vp = v + (size_t)b * SEQ * DMODEL + h * HSZ;

  v16bf aQ0 = load_a_rm(qp, DMODEL, rowBase, 0, lane);
  v16bf aQ1 = load_a_rm(qp, DMODEL, rowBase, 32, lane);

  v8f zero = {};
  v8f Oacc[4];
#pragma unroll
  for (int eb = 0; eb < 4; ++eb) Oacc[eb] = zero;
  float m[8], l[8];
#pragma unroll
  for (int r = 0; r < 8; ++r) { m[r] = -1e30f; l[r] = 0.f; }

  bf16* Pw = Pt[w];
  int nkt = q0 / 64 + 2;            // causal: key tiles up to block's max row
  for (int t = 0; t < nkt; ++t) {
    int kt0 = t * 64;
    __syncthreads();
    {   // cooperative K/V tile load: 64x64 bf16 each; V transposed on store
      int s0 = tid >> 2;
      int e0 = (tid & 3) * 16;
      const bf16* kg = kp + (size_t)(kt0 + s0) * DMODEL + e0;
      const bf16* vg = vp + (size_t)(kt0 + s0) * DMODEL + e0;
      bf16* ks = Kt + s0 * 64 + e0;
#pragma unroll
      for (int j = 0; j < 16; ++j) {
        ks[j] = kg[j];
        VtT[(e0 + j) * 64 + s0] = vg[j];
      }
    }
    __syncthreads();

    // S = Q K^T  (16 x 64), HS=64 split into two K=32 WMMA steps
    v8f S[4];
#pragma unroll
    for (int cb = 0; cb < 4; ++cb) S[cb] = zero;
#pragma unroll
    for (int cb = 0; cb < 4; ++cb) {
      v16bf b0 = load_b_T(Kt, 64, cb * 16, 0, lane);
      S[cb] = wmma_bf16(aQ0, b0, S[cb]);
      v16bf b1 = load_b_T(Kt, 64, cb * 16, 32, lane);
      S[cb] = wmma_bf16(aQ1, b1, S[cb]);
    }

    // scale + causal mask in place, track row max
    float tm[8];
#pragma unroll
    for (int r = 0; r < 8; ++r) tm[r] = -1e30f;
#pragma unroll
    for (int cb = 0; cb < 4; ++cb) {
      int col = kt0 + cb * 16 + (lane & 15);
#pragma unroll
      for (int r = 0; r < 8; ++r) {
        int row = rowBase + r + 8 * hi;
        float sv = S[cb][r] * 0.125f;   // 1/sqrt(64)
        if (col > row) sv = -1e30f;
        S[cb][r] = sv;
        tm[r] = fmaxf(tm[r], sv);
      }
    }
#pragma unroll
    for (int off = 1; off < 16; off <<= 1)
#pragma unroll
      for (int r = 0; r < 8; ++r)
        tm[r] = fmaxf(tm[r], __shfl_xor(tm[r], off, 32));

    float corr[8], rsum[8];
#pragma unroll
    for (int r = 0; r < 8; ++r) {
      float mn = fmaxf(m[r], tm[r]);
      corr[r] = __expf(m[r] - mn);
      m[r] = mn;
      rsum[r] = 0.f;
    }
    // p = exp(S - m): accumulate row sum and write bf16 P directly to LDS
#pragma unroll
    for (int cb = 0; cb < 4; ++cb) {
      int col = cb * 16 + (lane & 15);
#pragma unroll
      for (int r = 0; r < 8; ++r) {
        float p = __expf(S[cb][r] - m[r]);
        rsum[r] += p;
        Pw[(r + 8 * hi) * 64 + col] = to_bf16(p);
      }
    }
#pragma unroll
    for (int off = 1; off < 16; off <<= 1)
#pragma unroll
      for (int r = 0; r < 8; ++r)
        rsum[r] += __shfl_xor(rsum[r], off, 32);
#pragma unroll
    for (int r = 0; r < 8; ++r) l[r] = l[r] * corr[r] + rsum[r];
#pragma unroll
    for (int eb = 0; eb < 4; ++eb)
#pragma unroll
      for (int r = 0; r < 8; ++r) Oacc[eb][r] *= corr[r];

    // O += P * V via WMMA (P and V^T fragments are contiguous LDS loads)
#pragma unroll
    for (int kk = 0; kk < 2; ++kk) {
      v16bf aP = load_a_rm(Pw, 64, 0, kk * 32, lane);
#pragma unroll
      for (int eb = 0; eb < 4; ++eb) {
        v16bf bV = load_b_T(VtT, 64, eb * 16, kk * 32, lane);
        Oacc[eb] = wmma_bf16(aP, bV, Oacc[eb]);
      }
    }
  }

  // finalize: O / l, store bf16 at [b*T + row][h*64 + col]
  bf16* op = o + (size_t)b * SEQ * DMODEL + h * HSZ;
#pragma unroll
  for (int eb = 0; eb < 4; ++eb)
#pragma unroll
    for (int r = 0; r < 8; ++r) {
      int row = rowBase + r + 8 * hi;
      int col = eb * 16 + (lane & 15);
      op[(size_t)row * DMODEL + col] = to_bf16(Oacc[eb][r] / l[r]);
    }
}

// ---------------------------------------------------------------------------
extern "C" void kernel_launch(void* const* d_in, const int* in_sizes, int n_in,
                              void* d_out, int out_size, void* d_ws, size_t ws_size,
                              hipStream_t stream) {
  (void)in_sizes; (void)n_in; (void)out_size; (void)ws_size;
  const float* x   = (const float*)d_in[0];
  const float* Wq  = (const float*)d_in[1];
  const float* Wk  = (const float*)d_in[2];
  const float* Wv  = (const float*)d_in[3];
  const float* Wo  = (const float*)d_in[4];
  const float* bo  = (const float*)d_in[5];
  const float* W1  = (const float*)d_in[6];
  const float* b1  = (const float*)d_in[7];
  const float* W2  = (const float*)d_in[8];
  const float* b2  = (const float*)d_in[9];
  const float* g1  = (const float*)d_in[10];
  const float* be1 = (const float*)d_in[11];
  const float* g2  = (const float*)d_in[12];
  const float* be2 = (const float*)d_in[13];
  float* out = (float*)d_out;

  // workspace layout (~88 MB)
  char* ws = (char*)d_ws;
  size_t off = 0;
  bf16* hn = (bf16*)(ws + off); off += (size_t)NROWS * DMODEL * sizeof(bf16);
  bf16* qb = (bf16*)(ws + off); off += (size_t)NROWS * DMODEL * sizeof(bf16);
  bf16* kb = (bf16*)(ws + off); off += (size_t)NROWS * DMODEL * sizeof(bf16);
  bf16* vb = (bf16*)(ws + off); off += (size_t)NROWS * DMODEL * sizeof(bf16);
  bf16* ab = (bf16*)(ws + off); off += (size_t)NROWS * DMODEL * sizeof(bf16);
  float* x2 = (float*)(ws + off); off += (size_t)NROWS * DMODEL * sizeof(float);
  bf16* ff1 = (bf16*)(ws + off); off += (size_t)NROWS * DFF * sizeof(bf16);

  dim3 blk(256);
  dim3 gD(DMODEL / 128, NROWS / 128);   // N=1024 GEMMs
  dim3 gF(DFF / 128,    NROWS / 128);   // N=4096 GEMM

  // 1) LN1 -> hn (bf16)
  ln_kernel<<<NROWS, blk, 0, stream>>>(x, g1, be1, hn);
  // 2-4) Q/K/V projections (head-blocked weights [H][D][64])
  gemm_kernel<1, F_OUTB><<<gD, blk, 0, stream>>>(hn, Wq, nullptr, nullptr,
                                                 nullptr, qb, NROWS, DMODEL, DMODEL);
  gemm_kernel<1, F_OUTB><<<gD, blk, 0, stream>>>(hn, Wk, nullptr, nullptr,
                                                 nullptr, kb, NROWS, DMODEL, DMODEL);
  gemm_kernel<1, F_OUTB><<<gD, blk, 0, stream>>>(hn, Wv, nullptr, nullptr,
                                                 nullptr, vb, NROWS, DMODEL, DMODEL);
  // 5) causal flash attention
  attn_kernel<<<dim3(SEQ / 128, NBATCH * NHEAD), blk, 0, stream>>>(qb, kb, vb, ab);
  // 6) O-projection + bias + residual -> x2 (fp32)
  gemm_kernel<0, F_BIAS | F_RESID | F_OUTF><<<gD, blk, 0, stream>>>(
      ab, Wo, bo, x, x2, nullptr, NROWS, DMODEL, DMODEL);
  // 7) LN2 -> hn (reused, bf16)
  ln_kernel<<<NROWS, blk, 0, stream>>>(x2, g2, be2, hn);
  // 8) MLP up + ReLU -> ff1 (bf16)
  gemm_kernel<0, F_BIAS | F_RELU | F_OUTB><<<gF, blk, 0, stream>>>(
      hn, W1, b1, nullptr, nullptr, ff1, NROWS, DFF, DMODEL);
  // 9) MLP down + bias + residual -> out (fp32)
  gemm_kernel<0, F_BIAS | F_RESID | F_OUTF><<<gD, blk, 0, stream>>>(
      ff1, W2, b2, x2, out, nullptr, NROWS, DMODEL, DFF);
}
